// SuperONN2d_91113436217931
// MI455X (gfx1250) — matrix operand order repaired and checked by reading, not compiled, not run on hardware
//
#include <hip/hip_runtime.h>

// SuperONN2d fused forward for gfx1250 (MI455X).
//   x[16,64,128,128] f32, weight[64,192,3,3] f32, bias[64], shifts[64,2]
//   out[16,64,128,128] f32
// Fully fused: per-channel bilinear shift + powers (x,x^2,x^3) computed on the
// fly into LDS; 3x3 conv done as implicit GEMM with v_wmma_f32_16x16x32_bf16
// (bf16 operands, fp32 accumulation). Memory-bound: ~0.54 GB total HBM traffic.
// Round 3: async global->LDS weight staging with corrected builtin signature
// (generic v4i* params, per hipcc diagnostic) + next-slab global_prefetch.

typedef __attribute__((ext_vector_type(16))) __bf16 v16bf;
typedef __attribute__((ext_vector_type(8)))  __bf16 v8bf;
typedef __attribute__((ext_vector_type(8)))  float  v8f;
typedef __attribute__((__vector_size__(4 * sizeof(int)))) int v4i_t;

#define HH 128
#define WW 128
#define CIN 64
#define COUT 64
#define KD 192            // Q * CIN = 3 * 64
#define MAXSH 4.0f

#if defined(__AMDGCN__) && __has_builtin(__builtin_amdgcn_global_load_async_to_lds_b128)
#define USE_ASYNC_LDS 1
#endif

__device__ __forceinline__ void wait_async_lds() {
#if defined(USE_ASYNC_LDS)
#if __has_builtin(__builtin_amdgcn_s_wait_asynccnt)
  __builtin_amdgcn_s_wait_asynccnt(0);
#else
  asm volatile("s_wait_asynccnt 0x0" ::: "memory");
#endif
#endif
}

// ---------------------------------------------------------------------------
// Prep: weight[o][i][kh][kw] (OIHW f32) -> wb[(kh*3+kw)][o][i] bf16
// 9 tap-major [64 x 192] matrices, contiguous K per row (GEMM A-operand ready)
// ---------------------------------------------------------------------------
__global__ void superonn_prep_weights(const float* __restrict__ w,
                                      __bf16* __restrict__ wb) {
  int idx = blockIdx.x * blockDim.x + threadIdx.x;
  const int total = COUT * KD * 9;
  if (idx >= total) return;
  int kw = idx % 3; int t = idx / 3;
  int kh = t % 3;   t /= 3;
  int i  = t % KD;  int o = t / KD;
  float v = w[((o * KD + i) * 3 + kh) * 3 + kw];
  wb[((size_t)(kh * 3 + kw) * COUT + o) * KD + i] = (__bf16)v;
}

// ---------------------------------------------------------------------------
// Fused shift + powers + implicit-GEMM conv.
// Block: 256 threads (8 wave32). Tile: M=64 (co) x N=64 (w), K=192*9.
// Grid: (2 half-rows, 128 h, 16 n).
// ---------------------------------------------------------------------------
__launch_bounds__(256)
__global__ void superonn_fused(const float* __restrict__ x,
                               const __bf16* __restrict__ wb,
                               const float* __restrict__ bias,
                               const float* __restrict__ shifts,
                               float* __restrict__ out) {
  __shared__ __bf16 ldsX[66 * KD];     // xq row: [pos 0..65][k 0..191]
  __shared__ __bf16 ldsW[COUT * KD];   // weights one tap: [co][k]
  __shared__ float  s_sx[CIN];
  __shared__ float  s_sy[CIN];

  const int tid  = threadIdx.x;
  const int lane = tid & 31;
  const int wave = tid >> 5;
  const int n     = blockIdx.z;
  const int h     = blockIdx.y;
  const int wbase = blockIdx.x * 64;

  if (tid < CIN) {
    s_sx[tid] = shifts[tid * 2 + 0] * MAXSH;   // pixels along width
    s_sy[tid] = shifts[tid * 2 + 1] * MAXSH;   // pixels along height
  }

  // wave -> 2 tiles: mb selects co-block, nb0/nb0+1 select w-blocks
  const int mb  = wave & 3;
  const int nb0 = (wave >> 2) * 2;
  v8f acc0 = {};
  v8f acc1 = {};

  const int hiA   = (lane >= 16) ? 8 : 0;    // accumulator M-row offset (+8 for lanes 16..31)
  const int hiB   = (lane >= 16) ? 16 : 0;   // 16-bit B fragment K-half select
  const int rowA  = mb * 16 + (lane & 15);   // co row for A fragment

  for (int kh = 0; kh < 3; ++kh) {
    const int y = h + kh - 1;                // xq row (conv pad: OOB -> zeros)
    __syncthreads();                         // prior GEMM done reading ldsX

    // ---- stage one row of xq (shift + bilinear + powers) into LDS ----
    for (int idx = tid; idx < CIN * 66; idx += 256) {
      const int c  = idx / 66;
      const int wp = idx % 66;
      const int j  = wbase - 1 + wp;         // xs column (conv pad at edges)
      float v = 0.0f;
      if ((unsigned)y < (unsigned)HH && (unsigned)j < (unsigned)WW) {
        const float ys = (float)y + s_sy[c];
        const float xs = (float)j + s_sx[c];
        const float y0f = floorf(ys), x0f = floorf(xs);
        const float fy = ys - y0f, fx = xs - x0f;
        const int y0 = (int)y0f, x0 = (int)x0f;
        const float* plane = x + (size_t)(n * CIN + c) * (HH * WW);
        float g00 = 0.f, g01 = 0.f, g10 = 0.f, g11 = 0.f;
        const bool vy0 = (unsigned)y0 < (unsigned)HH;
        const bool vy1 = (unsigned)(y0 + 1) < (unsigned)HH;
        const bool vx0 = (unsigned)x0 < (unsigned)WW;
        const bool vx1 = (unsigned)(x0 + 1) < (unsigned)WW;
        if (vy0 && vx0) g00 = plane[y0 * WW + x0];
        if (vy0 && vx1) g01 = plane[y0 * WW + x0 + 1];
        if (vy1 && vx0) g10 = plane[(y0 + 1) * WW + x0];
        if (vy1 && vx1) g11 = plane[(y0 + 1) * WW + x0 + 1];
        v = (1.f - fy) * ((1.f - fx) * g00 + fx * g01)
          +         fy * ((1.f - fx) * g10 + fx * g11);
      }
      const float v2 = v * v;
      const float v3 = v2 * v;
      __bf16* row = &ldsX[wp * KD];
      row[c]           = (__bf16)v;    // q=1 channels   0..63
      row[CIN + c]     = (__bf16)v2;   // q=2 channels  64..127
      row[2 * CIN + c] = (__bf16)v3;   // q=3 channels 128..191
    }

    for (int kw = 0; kw < 3; ++kw) {
      const int tap = kh * 3 + kw;
      __syncthreads();   // ldsX ready / prior GEMM done reading ldsW

      // ---- stage A-slab of weights for this tap ----
      {
        const uint4* src = (const uint4*)(wb + (size_t)tap * COUT * KD);
        uint4* dst = (uint4*)ldsW;
        const int nvec = (COUT * KD * 2) / 16;     // 1536 x 16B
#if defined(USE_ASYNC_LDS)
        for (int i2 = tid; i2 < nvec; i2 += 256) {
          __builtin_amdgcn_global_load_async_to_lds_b128(
              (v4i_t*)(src + i2), (v4i_t*)(dst + i2), 0, 0);
        }
        wait_async_lds();
#else
        for (int i2 = tid; i2 < nvec; i2 += 256) dst[i2] = src[i2];
#endif
      }
      __syncthreads();

      // prefetch next tap's weight slab into L2/L0 while GEMM runs
      if (tap < 8 && tid < (COUT * KD * 2) / 128) {
        __builtin_prefetch(((const char*)(wb + (size_t)(tap + 1) * COUT * KD))
                               + tid * 128, 0, 1);
      }

      // ---- GEMM: M=co, N=w, K=192 in 6 steps of 32 ----
      const int col0 = nb0 * 16 + (lane & 15) + kw;   // pos into 66-wide row
      const int col1 = col0 + 16;
      #pragma unroll
      for (int kc = 0; kc < KD; kc += 32) {
        // A fragment (weights): K = kc + hiB + {0..15}
        v8bf alo = *(const v8bf*)&ldsW[rowA * KD + kc + hiB];
        v8bf ahi = *(const v8bf*)&ldsW[rowA * KD + kc + hiB + 8];
        v16bf a = __builtin_shufflevector(alo, ahi,
                    0, 1, 2, 3, 4, 5, 6, 7, 8, 9, 10, 11, 12, 13, 14, 15);
        // B fragments (xq): K = kc + hiB + {0..15}, contiguous in LDS
        v16bf b0 = *(const v16bf*)&ldsX[col0 * KD + kc + hiB];
        v16bf b1 = *(const v16bf*)&ldsX[col1 * KD + kc + hiB];
        acc0 = __builtin_amdgcn_wmma_f32_16x16x32_bf16(
                   false, a, false, b0, (short)0, acc0, false, false);
        acc1 = __builtin_amdgcn_wmma_f32_16x16x32_bf16(
                   false, a, false, b1, (short)0, acc1, false, false);
      }
    }
  }

  // ---- epilogue: bias + coalesced stores (N = w runs along lanes 0..15) ----
  const int w0 = wbase + nb0 * 16 + (lane & 15);
  #pragma unroll
  for (int r = 0; r < 8; ++r) {
    const int coc = mb * 16 + r + hiA;       // M = vgpr + 8*(lane>=16)
    const float bi = bias[coc];
    const size_t o0 = (((size_t)n * COUT + coc) * HH + h) * WW;
    out[o0 + w0]      = acc0[r] + bi;
    out[o0 + w0 + 16] = acc1[r] + bi;
  }
}

// ---------------------------------------------------------------------------
extern "C" void kernel_launch(void* const* d_in, const int* in_sizes, int n_in,
                              void* d_out, int out_size, void* d_ws, size_t ws_size,
                              hipStream_t stream) {
  const float* x      = (const float*)d_in[0];
  const float* weight = (const float*)d_in[1];
  const float* bias   = (const float*)d_in[2];
  const float* shifts = (const float*)d_in[3];
  float* out = (float*)d_out;

  __bf16* wb = (__bf16*)d_ws;   // 9 * 64 * 192 bf16 = 216 KB

  const int wtotal = COUT * KD * 9;
  superonn_prep_weights<<<(wtotal + 255) / 256, 256, 0, stream>>>(weight, wb);

  dim3 grid(WW / 64, HH, 16);   // (half-row, h, n) = 4096 blocks
  superonn_fused<<<grid, 256, 0, stream>>>(x, wb, bias, shifts, out);
}
